// GINEncoder_5677946765458
// MI455X (gfx1250) — matrix operand order repaired and checked by reading, not compiled, not run on hardware
//
#include <hip/hip_runtime.h>
#include <hip/hip_bf16.h>

// GIN encoder, 3 layers. Shapes fixed by the reference.
#define N_NODESC 50000
#define N_EDGESC 800000
#define DIMC     64
#define MIDC     128
#define BN_EPSC  1e-5f

typedef __attribute__((ext_vector_type(16))) __bf16 v16bf;
typedef __attribute__((ext_vector_type(8)))  float  v8f;

union ABu { v16bf v; uint4 q[2]; };

// ---------------- utility ----------------
__global__ void k_zero(float* __restrict__ p, long n) {
  long i = (long)blockIdx.x * blockDim.x + threadIdx.x;
  long st = (long)gridDim.x * blockDim.x;
  for (; i < n; i += st) p[i] = 0.f;
}

// ---------------- edge scatter-add:  agg[dst] += h[src]  ----------------
// thread = (edge, feature); 64 features contiguous -> coalesced loads & atomics.
__global__ void k_scatter(const float* __restrict__ h, const int* __restrict__ src,
                          const int* __restrict__ dst, float* __restrict__ agg, int nedges) {
  long t = (long)blockIdx.x * blockDim.x + threadIdx.x;
  int e = (int)(t >> 6);
  int f = (int)(t & 63);
  if (e >= nedges) return;
  int s = src[e];
  int d = dst[e];
  unsafeAtomicAdd(&agg[(long)d * DIMC + f], h[(long)s * DIMC + f]);  // global_atomic_add_f32
}

// ---------------- z = (1+eps)*h + agg, emitted as bf16 GEMM input ----------------
__global__ void k_make_z(const float* __restrict__ h, const float* __restrict__ agg,
                         const float* __restrict__ epsArr, int layer,
                         __bf16* __restrict__ zb, long n) {
  float e1 = 1.f + epsArr[layer];
  long i = (long)blockIdx.x * blockDim.x + threadIdx.x;
  long st = (long)gridDim.x * blockDim.x;
  for (; i < n; i += st) zb[i] = (__bf16)(e1 * h[i] + agg[i]);
}

// ---------------- pack row-major f32 weights (K x NC) into WMMA B layout ----------------
// B tile (32x16 bf16): lane = {half, col}; vector elem i -> K = kc*32 + half*8 + i (i<8), kc*32 + 16 + half*8 + (i-8) (i>=8); col = tn*16+l16.
// Packed contiguous as [tn][kc][lane][16 elems] so GEMM loads 32B/lane.
__global__ void k_pack_w(const float* __restrict__ w, __bf16* __restrict__ wp, int K, int NC) {
  int t = blockIdx.x * blockDim.x + threadIdx.x;
  int kChunks = K >> 5;
  int total = (NC >> 4) * kChunks * 512;
  if (t >= total) return;
  int i    = t & 15;
  int lane = (t >> 4) & 31;
  int kcTn = t >> 9;
  int kc = kcTn % kChunks;
  int tn = kcTn / kChunks;
  int half = lane >> 4, l16 = lane & 15;
  int Kidx = kc * 32 + (i < 8 ? half * 8 + i : 16 + half * 8 + (i - 8));
  int col  = tn * 16 + l16;
  wp[t] = (__bf16)w[(long)Kidx * NC + col];
}

// ---------------- strip GEMM with fused BN column stats ----------------
// out(f32 N x NC) = A(bf16 N x K, row-major) @ Bp + bias;  s/ss += per-column sum/sumsq.
// One wave: 16 rows x (TNW*16) cols. A strip held in registers across the column sweep.
// KCH = K/32 chunks, TNT = NC/16 total col tiles, TNW = col tiles per wave.
template <int KCH, int TNT, int TNW>
__global__ void k_gemm_strip(const __bf16* __restrict__ A, const __bf16* __restrict__ Bp,
                             const float* __restrict__ bias, float* __restrict__ out,
                             float* __restrict__ s, float* __restrict__ ss, int nrows) {
  constexpr int K  = KCH * 32;
  constexpr int NC = TNT * 16;
  constexpr int GROUPS = TNT / TNW;
  int wave = (int)(((long)blockIdx.x * blockDim.x + threadIdx.x) >> 5);
  int lane = threadIdx.x & 31;
  int tilesM = nrows >> 4;
  if (wave >= tilesM * GROUPS) return;  // wave-uniform exit: EXEC stays all-ones for WMMA
  int tm = wave / GROUPS;
  int tg = wave % GROUPS;
  int half = lane >> 4, l16 = lane & 15;

  // A 16x32 bf16 layout: per lane two contiguous 8-elem runs at kb and kb+16
  const __bf16* Arow = A + (long)(tm * 16 + l16) * K;
  ABu a[KCH];
#pragma unroll
  for (int kc = 0; kc < KCH; ++kc) {
    int kb = kc * 32 + half * 8;
    a[kc].q[0] = *reinterpret_cast<const uint4*>(Arow + kb);
    a[kc].q[1] = *reinterpret_cast<const uint4*>(Arow + kb + 16);
  }

  int rowbase = tm * 16 + half * 8;
#pragma unroll
  for (int t = 0; t < TNW; ++t) {
    int tn = tg * TNW + t;
    v8f acc = {};
#pragma unroll
    for (int kc = 0; kc < KCH; ++kc) {
      ABu b;
      const __bf16* bp = Bp + (((long)tn * KCH + kc) * 32 + lane) * 16;
      b.q[0] = *reinterpret_cast<const uint4*>(bp);
      b.q[1] = *reinterpret_cast<const uint4*>(bp + 8);
      acc = __builtin_amdgcn_wmma_f32_16x16x32_bf16(false, a[kc].v, false, b.v,
                                                    (short)0, acc, false, false);
    }
    // C/D layout: VGPR r, lanes0-15 -> M=r, N=lane; lanes16-31 -> M=r+8
    int col = tn * 16 + l16;
    float bc = bias[col];
    float cs = 0.f, css = 0.f;
#pragma unroll
    for (int r = 0; r < 8; ++r) {
      float v = acc[r] + bc;
      out[(long)(rowbase + r) * NC + col] = v;
      cs += v; css += v * v;
    }
    // lanes l and l+16 hold the same column (rows 0-7 / 8-15): pairwise combine
    cs  += __shfl_xor(cs, 16);
    css += __shfl_xor(css, 16);
    if (half == 0) {
      unsafeAtomicAdd(&s[col], cs);
      unsafeAtomicAdd(&ss[col], css);
    }
  }
}

// ---------------- BN(normalize)+ReLU, bf16 output (feeds next GEMM) ----------------
__global__ void k_norm_relu_bf(const float* __restrict__ y, const float* __restrict__ s,
                               const float* __restrict__ ss, const float* __restrict__ g,
                               const float* __restrict__ be, __bf16* __restrict__ out,
                               long nrows, int ncols) {
  __shared__ float sc[MIDC], sh[MIDC];
  if ((int)threadIdx.x < ncols) {
    int c = threadIdx.x;
    float m = s[c] / (float)nrows;
    float v = ss[c] / (float)nrows - m * m;
    float k = g[c] * rsqrtf(v + BN_EPSC);
    sc[c] = k; sh[c] = be[c] - m * k;
  }
  __syncthreads();
  long n = nrows * ncols;
  long i = (long)blockIdx.x * blockDim.x + threadIdx.x;
  long st = (long)gridDim.x * blockDim.x;
  for (; i < n; i += st) {
    int c = (int)(i % ncols);
    float v = fmaf(y[i], sc[c], sh[c]);
    out[i] = (__bf16)fmaxf(v, 0.f);
  }
}

// ---------------- BN+ReLU, f32 output, fused accumulation of outer-BN stats ----------------
__global__ void k_norm_relu_stats(const float* __restrict__ y, const float* __restrict__ s,
                                  const float* __restrict__ ss, const float* __restrict__ g,
                                  const float* __restrict__ be, float* __restrict__ out,
                                  float* __restrict__ s2, float* __restrict__ ss2,
                                  int nrows, int ncols, int chunk) {
  int c = threadIdx.x;               // blockDim.x == ncols
  float m = s[c] / (float)nrows;
  float v = ss[c] / (float)nrows - m * m;
  float k = g[c] * rsqrtf(v + BN_EPSC);
  float sh = be[c] - m * k;
  int r0 = blockIdx.x * chunk;
  int r1 = min(r0 + chunk, nrows);
  float a = 0.f, b = 0.f;
  for (int r = r0; r < r1; ++r) {
    float z = fmaxf(fmaf(y[(long)r * ncols + c], k, sh), 0.f);
    out[(long)r * ncols + c] = z;
    a += z; b += z * z;
  }
  unsafeAtomicAdd(&s2[c], a);
  unsafeAtomicAdd(&ss2[c], b);
}

// ---------------- outer BN (affine only), f32 output ----------------
__global__ void k_norm_affine(const float* __restrict__ y, const float* __restrict__ s,
                              const float* __restrict__ ss, const float* __restrict__ g,
                              const float* __restrict__ be, float* __restrict__ out,
                              long nrows, int ncols) {
  __shared__ float sc[DIMC], sh[DIMC];
  if ((int)threadIdx.x < ncols) {
    int c = threadIdx.x;
    float m = s[c] / (float)nrows;
    float v = ss[c] / (float)nrows - m * m;
    float k = g[c] * rsqrtf(v + BN_EPSC);
    sc[c] = k; sh[c] = be[c] - m * k;
  }
  __syncthreads();
  long n = nrows * ncols;
  long i = (long)blockIdx.x * blockDim.x + threadIdx.x;
  long st = (long)gridDim.x * blockDim.x;
  for (; i < n; i += st) {
    int c = (int)(i % ncols);
    out[i] = fmaf(y[i], sc[c], sh[c]);
  }
}

// ==================================================================================
extern "C" void kernel_launch(void* const* d_in, const int* in_sizes, int n_in,
                              void* d_out, int out_size, void* d_ws, size_t ws_size,
                              hipStream_t stream) {
  const int N = N_NODESC, E = N_EDGESC;
  // d_in order: x, edge_index, then 3 layers of {w1,b1,g1,be1,w2,b2,g2,be2,gn,bn} (dict
  // insertion order in _make_layer), then eps[3].
  const float* x  = (const float*)d_in[0];
  const int*   ei = (const int*)d_in[1];          // [2, E]; src = row0, dst = row1
  const int*   src = ei;
  const int*   dst = ei + E;
  const float* P[3][10];
  for (int l = 0; l < 3; ++l)
    for (int j = 0; j < 10; ++j) P[l][j] = (const float*)d_in[2 + l * 10 + j];
  const float* epsArr = (const float*)d_in[2 + 30];

  // ---- workspace carve-up (all 256B aligned) ----
  char* w = (char*)d_ws;
  size_t off = 0;
  auto take = [&](size_t bytes) { void* p = w + off; off = (off + bytes + 255) & ~(size_t)255; return p; };
  float*  agg   = (float*) take((size_t)N * DIMC * 4);
  __bf16* zb    = (__bf16*)take((size_t)N * DIMC * 2);
  float*  y1    = (float*) take((size_t)N * MIDC * 4);
  __bf16* y1n   = (__bf16*)take((size_t)N * MIDC * 2);
  float*  y2    = (float*) take((size_t)N * DIMC * 4);
  float*  z2    = (float*) take((size_t)N * DIMC * 4);
  float*  hbuf0 = (float*) take((size_t)N * DIMC * 4);
  float*  hbuf1 = (float*) take((size_t)N * DIMC * 4);
  __bf16* w1p[3]; __bf16* w2p[3];
  for (int l = 0; l < 3; ++l) { w1p[l] = (__bf16*)take(DIMC * MIDC * 2); w2p[l] = (__bf16*)take(MIDC * DIMC * 2); }
  float* stats = (float*)take(512 * 4);
  float *s1 = stats, *ss1 = stats + 128, *s2 = stats + 256, *ss2 = stats + 320,
        *s3 = stats + 384, *ss3 = stats + 448;

  // ---- pack all weights to WMMA layout (tiny, once per launch) ----
  for (int l = 0; l < 3; ++l) {
    k_pack_w<<<(DIMC / 32) * (MIDC / 16) * 2, 256, 0, stream>>>(P[l][0], w1p[l], DIMC, MIDC);
    k_pack_w<<<(MIDC / 32) * (DIMC / 16) * 2, 256, 0, stream>>>(P[l][4], w2p[l], MIDC, DIMC);
  }

  const int CHUNK = 64;
  const int nchunk = (N + CHUNK - 1) / CHUNK;
  const long nd = (long)N * DIMC;
  const int tilesM = N / 16;  // 3125, exact

  const float* hin = x;
  for (int l = 0; l < 3; ++l) {
    float* hout = (l == 0) ? hbuf0 : (l == 1) ? hbuf1 : (float*)d_out;

    k_zero<<<2048, 256, 0, stream>>>(agg, nd);
    k_zero<<<2, 256, 0, stream>>>(stats, 512);

    k_scatter<<<(int)(((long)E * 64 + 255) / 256), 256, 0, stream>>>(hin, src, dst, agg, E);
    k_make_z<<<2048, 256, 0, stream>>>(hin, agg, epsArr, l, zb, nd);

    // Linear1 (64->128): K=64 (KCH=2), NC=128 (TNT=8), 4 col tiles per wave
    {
      int waves = tilesM * (8 / 4);
      k_gemm_strip<2, 8, 4><<<(waves * 32 + 255) / 256, 256, 0, stream>>>(
          zb, w1p[l], P[l][1], y1, s1, ss1, N);
    }
    k_norm_relu_bf<<<2048, 256, 0, stream>>>(y1, s1, ss1, P[l][2], P[l][3], y1n, N, MIDC);

    // Linear2 (128->64): K=128 (KCH=4), NC=64 (TNT=4), 2 col tiles per wave
    {
      int waves = tilesM * (4 / 2);
      k_gemm_strip<4, 4, 2><<<(waves * 32 + 255) / 256, 256, 0, stream>>>(
          y1n, w2p[l], P[l][5], y2, s2, ss2, N);
    }
    k_norm_relu_stats<<<nchunk, DIMC, 0, stream>>>(y2, s2, ss2, P[l][6], P[l][7], z2, s3, ss3, N, DIMC, CHUNK);

    // outer BatchNorm
    k_norm_affine<<<2048, 256, 0, stream>>>(z2, s3, ss3, P[l][8], P[l][9], hout, N, DIMC);

    hin = hout;
  }
  (void)n_in; (void)in_sizes; (void)out_size; (void)ws_size;
}
